// AttentionFusion_44573170598091
// MI455X (gfx1250) — matrix operand order repaired and checked by reading, not compiled, not run on hardware
//
#include <hip/hip_runtime.h>
#include <hip/hip_bf16.h>

typedef __bf16 bf16t;
typedef __attribute__((ext_vector_type(16))) __bf16 v16bf;
typedef __attribute__((ext_vector_type(8)))  float  v8f;
typedef __attribute__((ext_vector_type(4)))  unsigned int u32x4;

#define NTHREADS 256
#define NWAVES   8
#define KBM      16   // batch rows per workgroup

union FragU { u32x4 u[2]; v16bf v; };
union C8   { u32x4 u; bf16t h[8]; };       // 8 bf16 = 16B chunk
union F4   { u32x4 u; float f[4]; };       // 4 f32  = 16B chunk
union H4   { unsigned long long q; bf16t h[4]; };

// ---- explicit global-address-space accesses (global_*, not flat_*) ----------
__device__ __forceinline__ u32x4 gload4(const void* p) {
  return *(__attribute__((address_space(1))) const u32x4*)p;
}
__device__ __forceinline__ float gloadf(const float* p) {
  return *(__attribute__((address_space(1))) const float*)p;
}
__device__ __forceinline__ void gstore8(void* p, unsigned long long v) {
  *(__attribute__((address_space(1))) unsigned long long*)p = v;
}
__device__ __forceinline__ C8 glb_c8(const bf16t* p) {
  C8 c; c.u = gload4(p); return c;
}
__device__ __forceinline__ C8 lds_c8(const bf16t* p) {
  C8 c; c.u = *(const u32x4*)p; return c;
}

// A fragment (16x32 bf16, M x K), LDS K-contiguous rows of `pitch` elements.
// ISA layout: lanes 0-15 = M rows, VGPR0-3 <- K[8h..8h+7], VGPR4-7 <- K[16+8h..+7].
__device__ __forceinline__ v16bf lds_loadA(const bf16t* base, int pitch) {
  int lane = threadIdx.x & 31;
  int lh = lane >> 4;
  int m  = lane & 15;
  const bf16t* p = base + m * pitch + 8 * lh;
  FragU f;
  f.u[0] = *(const u32x4*)(p);
  f.u[1] = *(const u32x4*)(p + 16);
  return f.v;
}

// B fragment (32x16 bf16, K x N) from weights stored torch-style (N,K) row major.
// ISA layout: lane n holds column n; VGPR0-3 <- K[8h..8h+7], VGPR4-7 <- K[16+8h..+7].
__device__ __forceinline__ v16bf glb_loadB(const bf16t* W, int ldw, int n0, int k0) {
  int lane = threadIdx.x & 31;
  int lh = lane >> 4;
  int n  = lane & 15;
  const bf16t* p = W + (size_t)(n0 + n) * ldw + k0 + 8 * lh;
  FragU f;
  f.u[0] = gload4(p);
  f.u[1] = gload4(p + 16);
  return f.v;
}

__device__ __forceinline__ v8f wmma_bf16(v16bf a, v16bf b, v8f c) {
  return __builtin_amdgcn_wmma_f32_16x16x32_bf16(false, a, false, b, (short)0, c,
                                                 false, false);
}

// Double-buffered multi-tile GEMM: one wave computes NT 16x16 C tiles
// (columns n0..n0+16*NT) against a shared A row-block. Two explicit fragment
// sets alternate so refills land in dead registers and global/LDS loads
// overlap the WMMA issue. Requires K % 64 == 0.
template <int NT, typename FA>
__device__ __forceinline__ void gemm_rowblock(FA aAddr, int apitch,
                                              const bf16t* W, int ldw,
                                              int n0, int K, v8f* acc) {
  v16bf a0 = lds_loadA(aAddr(0), apitch);
  v16bf b0[NT];
  #pragma unroll
  for (int j = 0; j < NT; ++j) b0[j] = glb_loadB(W, ldw, n0 + 16 * j, 0);
  v16bf a1 = lds_loadA(aAddr(32), apitch);
  v16bf b1[NT];
  #pragma unroll
  for (int j = 0; j < NT; ++j) b1[j] = glb_loadB(W, ldw, n0 + 16 * j, 32);

  for (int k0 = 64; k0 < K; k0 += 64) {
    #pragma unroll
    for (int j = 0; j < NT; ++j) acc[j] = wmma_bf16(a0, b0[j], acc[j]);
    a0 = lds_loadA(aAddr(k0), apitch);
    #pragma unroll
    for (int j = 0; j < NT; ++j) b0[j] = glb_loadB(W, ldw, n0 + 16 * j, k0);
    #pragma unroll
    for (int j = 0; j < NT; ++j) acc[j] = wmma_bf16(a1, b1[j], acc[j]);
    a1 = lds_loadA(aAddr(k0 + 32), apitch);
    #pragma unroll
    for (int j = 0; j < NT; ++j) b1[j] = glb_loadB(W, ldw, n0 + 16 * j, k0 + 32);
  }
  #pragma unroll
  for (int j = 0; j < NT; ++j) acc[j] = wmma_bf16(a0, b0[j], acc[j]);
  #pragma unroll
  for (int j = 0; j < NT; ++j) acc[j] = wmma_bf16(a1, b1[j], acc[j]);
}

__global__ void __launch_bounds__(NTHREADS)
cvt_bf16(const float* __restrict__ src, bf16t* __restrict__ dst, int n4) {
  int i = blockIdx.x * NTHREADS + threadIdx.x;
  if (i < n4) {
    F4 t; t.u = gload4(src + i * 4);
    H4 o;
    #pragma unroll
    for (int k = 0; k < 4; ++k) o.h[k] = (bf16t)t.f[k];
    gstore8(dst + i * 4, o.q);
  }
}

__global__ void __launch_bounds__(NTHREADS)
fusion_fwd(const float* __restrict__ rgb, const float* __restrict__ pose,
           const float* __restrict__ flow,
           const bf16t* Wr, const float* br, const float* gr, const float* betar,
           const bf16t* Wp, const float* bp, const float* gp, const float* betap,
           const bf16t* Wf, const float* bf_, const float* gf, const float* betaf,
           const bf16t* Win, const float* bin,
           const bf16t* Wou, const float* bout,
           const bf16t* Wg1, const float* bg1,
           const bf16t* Wg2, const float* bg2,
           const bf16t* Wc1, const float* bc1, const float* cg, const float* cbeta,
           const bf16t* Wc2, const float* bc2,
           float* __restrict__ out_logits, float* __restrict__ out_gate)
{
  extern __shared__ char smem[];
  // LDS map (bytes):
  //   sAct   [0, 49920)            : 48 x 520 bf16  (stacked r/p/f activations)
  //   qreg   [49920, 198144)       : 48 x 1544 bf16 (qkv); overlaid:
  //       sIn  @ +0      : 16 x (K+8) bf16 input staging (step 0 only)
  //       sTmp @ +36864  : 16 x 516 f32  GEMM scratch for LN (steps 0, 6)
  //       sAttO@ +98304  : 48 x 520 bf16 attended output (steps 3..5)
  //   sAtt   [198144, 248064)      : 48 x 520 bf16 attention context
  //   sGateH [248064, 264704)      : 16 x 520 bf16 gate hidden
  //   sFused [264704, 281344)      : 16 x 520 bf16 fused features
  //   sClsH  [281344, 289792)      : 16 x 264 bf16 classifier hidden
  //   sGate  [289792, 289984)      : 48 f32 gate weights
  bf16t* sAct   = (bf16t*)smem;
  char*  qreg   = smem + 49920;
  bf16t* sQKV   = (bf16t*)qreg;
  bf16t* sIn    = (bf16t*)qreg;
  float* sTmp   = (float*)(qreg + 36864);
  bf16t* sAttO  = (bf16t*)(qreg + 98304);
  bf16t* sAtt   = (bf16t*)(smem + 198144);
  bf16t* sGateH = (bf16t*)(smem + 248064);
  bf16t* sFused = (bf16t*)(smem + 264704);
  bf16t* sClsH  = (bf16t*)(smem + 281344);
  float* sGate  = (float*)(smem + 289792);

  const int tid  = threadIdx.x;
  const int wave = tid >> 5;
  const int lane = tid & 31;
  const int cn   = lane & 15;        // C tile: column
  const int cm   = (lane >> 4) * 8;  // C tile: row base (M 0..7 / 8..15)
  const int b0   = blockIdx.x * KBM;
  const v8f vz   = {0.f, 0.f, 0.f, 0.f, 0.f, 0.f, 0.f, 0.f};

  // ---------------- Step 0: per-modality Linear -> LN -> ReLU ----------------
  {
    const float* xs[3]  = {rgb, pose, flow};
    const bf16t* Wm[3]  = {Wr, Wp, Wf};
    const float* bm[3]  = {br, bp, bf_};
    const float* gm[3]  = {gr, gp, gf};
    const float* bem[3] = {betar, betap, betaf};
    const int    Km[3]  = {768, 640, 1024};
    for (int mod = 0; mod < 3; ++mod) {
      const int K   = Km[mod];
      const int K4  = K >> 2;
      const int pIn = K + 8;
      const float* x = xs[mod];
      // stage input tile as bf16 (vectorized: b128 load, b64 LDS store)
      for (int i = tid; i < KBM * K4; i += NTHREADS) {
        int r = i / K4, c4 = (i - r * K4) * 4;
        F4 t; t.u = gload4(x + (size_t)(b0 + r) * K + c4);
        H4 o;
        #pragma unroll
        for (int k = 0; k < 4; ++k) o.h[k] = (bf16t)t.f[k];
        *(unsigned long long*)(sIn + r * pIn + c4) = o.q;
      }
      __syncthreads();
      {
        int n0 = wave * 64;
        v8f acc[4] = {vz, vz, vz, vz};
        gemm_rowblock<4>([&](int k0) { return sIn + k0; }, pIn,
                         Wm[mod], K, n0, K, acc);
        #pragma unroll
        for (int jj = 0; jj < 4; ++jj) {
          int nc = n0 + jj * 16 + cn;
          float bias = gloadf(bm[mod] + nc);
          #pragma unroll
          for (int r = 0; r < 8; ++r)
            sTmp[(cm + r) * 516 + nc] = acc[jj][r] + bias;
        }
      }
      __syncthreads();
      // LayerNorm + ReLU over E=512, one wave handles two rows
      for (int rr = 0; rr < 2; ++rr) {
        int row = wave * 2 + rr;
        float v[16], s1 = 0.f, s2 = 0.f;
        #pragma unroll
        for (int i = 0; i < 16; ++i) {
          float t = sTmp[row * 516 + lane + 32 * i];
          v[i] = t; s1 += t; s2 += t * t;
        }
        #pragma unroll
        for (int m = 16; m >= 1; m >>= 1) {
          s1 += __shfl_xor(s1, m, 32);
          s2 += __shfl_xor(s2, m, 32);
        }
        float mean = s1 * (1.f / 512.f);
        float rstd = rsqrtf(s2 * (1.f / 512.f) - mean * mean + 1e-5f);
        bf16t* dst = sAct + (row * 3 + mod) * 520;
        #pragma unroll
        for (int i = 0; i < 16; ++i) {
          int e = lane + 32 * i;
          float y = (v[i] - mean) * rstd * gloadf(gm[mod] + e) +
                    gloadf(bem[mod] + e);
          dst[e] = (bf16t)fmaxf(y, 0.f);
        }
      }
      __syncthreads();
    }
  }

  // ---------------- Step 1: QKV = stacked @ in_w.T + in_b --------------------
  for (int mt = 0; mt < 3; ++mt) {
    int m0 = mt * 16;
    for (int gg = 0; gg < 3; ++gg) {
      int n0 = (wave + 8 * gg) * 64;   // 24 groups of 4 tiles cover N=1536
      v8f acc[4] = {vz, vz, vz, vz};
      gemm_rowblock<4>([&](int k0) { return sAct + m0 * 520 + k0; }, 520,
                       Win, 512, n0, 512, acc);
      #pragma unroll
      for (int jj = 0; jj < 4; ++jj) {
        int nc = n0 + jj * 16 + cn;
        float bias = gloadf(bin + nc);
        bf16t* dst = sQKV + (m0 + cm) * 1544 + nc;
        #pragma unroll
        for (int r = 0; r < 8; ++r) dst[r * 1544] = (bf16t)(acc[jj][r] + bias);
      }
    }
  }
  __syncthreads();

  // ---------------- Step 2: 3-token / 8-head attention (16B-chunked) ---------
  if (tid < KBM * 8) {
    int b = tid >> 3, h = tid & 7;
    const bf16t* base = sQKV + (b * 3) * 1544 + h * 64;
    float sc[3][3];
    #pragma unroll
    for (int i = 0; i < 3; ++i)
      #pragma unroll
      for (int j = 0; j < 3; ++j) {
        const bf16t* qi = base + i * 1544;
        const bf16t* kj = base + 512 + j * 1544;
        float s = 0.f;
        #pragma unroll
        for (int c = 0; c < 8; ++c) {
          C8 qc = lds_c8(qi + c * 8);
          C8 kc = lds_c8(kj + c * 8);
          #pragma unroll
          for (int e = 0; e < 8; ++e) s += (float)qc.h[e] * (float)kc.h[e];
        }
        sc[i][j] = s * 0.125f;  // 1/sqrt(64)
      }
    const bf16t* vb = base + 1024;
    #pragma unroll
    for (int i = 0; i < 3; ++i) {
      float mx = fmaxf(sc[i][0], fmaxf(sc[i][1], sc[i][2]));
      float e0 = __expf(sc[i][0] - mx);
      float e1 = __expf(sc[i][1] - mx);
      float e2 = __expf(sc[i][2] - mx);
      float inv = 1.f / (e0 + e1 + e2);
      float w0 = e0 * inv, w1 = e1 * inv, w2 = e2 * inv;
      bf16t* dst = sAtt + (b * 3 + i) * 520 + h * 64;
      #pragma unroll
      for (int c = 0; c < 8; ++c) {
        C8 v0 = lds_c8(vb + c * 8);
        C8 v1 = lds_c8(vb + 1544 + c * 8);
        C8 v2 = lds_c8(vb + 2 * 1544 + c * 8);
        C8 o;
        #pragma unroll
        for (int e = 0; e < 8; ++e)
          o.h[e] = (bf16t)(w0 * (float)v0.h[e] + w1 * (float)v1.h[e] +
                           w2 * (float)v2.h[e]);
        *(u32x4*)(dst + c * 8) = o.u;
      }
    }
  }
  __syncthreads();

  // ---------------- Step 3: out-projection -----------------------------------
  for (int mt = 0; mt < 3; ++mt) {
    int m0 = mt * 16;
    int n0 = wave * 64;
    v8f acc[4] = {vz, vz, vz, vz};
    gemm_rowblock<4>([&](int k0) { return sAtt + m0 * 520 + k0; }, 520,
                     Wou, 512, n0, 512, acc);
    #pragma unroll
    for (int jj = 0; jj < 4; ++jj) {
      int nc = n0 + jj * 16 + cn;
      float bias = gloadf(bout + nc);
      bf16t* dst = sAttO + (m0 + cm) * 520 + nc;
      #pragma unroll
      for (int r = 0; r < 8; ++r) dst[r * 520] = (bf16t)(acc[jj][r] + bias);
    }
  }

  // ---------------- Step 4: gate MLP -----------------------------------------
  // hidden = relu(concat(r,p,f) @ gw1.T + gb1); concat row b == sAct rows 3b..3b+2
  {
    int n0 = wave * 64;
    v8f acc[4] = {vz, vz, vz, vz};
    gemm_rowblock<4>(
        [&](int k0) { return sAct + (k0 >> 9) * 520 + (k0 & 511); }, 1560,
        Wg1, 1536, n0, 1536, acc);
    #pragma unroll
    for (int jj = 0; jj < 4; ++jj) {
      int nc = n0 + jj * 16 + cn;
      float bias = gloadf(bg1 + nc);
      bf16t* dst = sGateH + cm * 520 + nc;
      #pragma unroll
      for (int r = 0; r < 8; ++r)
        dst[r * 520] = (bf16t)fmaxf(acc[jj][r] + bias, 0.f);
    }
  }
  __syncthreads();
  if (tid < KBM * 3) {
    int b = tid / 3, s = tid - 3 * (tid / 3);
    float acc = gloadf(bg2 + s);
    const bf16t* hb = sGateH + b * 520;
    const bf16t* w  = Wg2 + s * 512;
    #pragma unroll 8
    for (int c = 0; c < 64; ++c) {
      C8 hc = lds_c8(hb + c * 8);
      C8 wc = glb_c8(w + c * 8);
      #pragma unroll
      for (int e = 0; e < 8; ++e) acc += (float)hc.h[e] * (float)wc.h[e];
    }
    sGate[b * 3 + s] = acc;
  }
  __syncthreads();
  if (tid < KBM) {
    int b = tid;
    float l0 = sGate[b * 3], l1 = sGate[b * 3 + 1], l2 = sGate[b * 3 + 2];
    float mx = fmaxf(l0, fmaxf(l1, l2));
    float e0 = __expf(l0 - mx), e1 = __expf(l1 - mx), e2 = __expf(l2 - mx);
    float inv = 1.f / (e0 + e1 + e2);
    sGate[b * 3 + 0] = e0 * inv;
    sGate[b * 3 + 1] = e1 * inv;
    sGate[b * 3 + 2] = e2 * inv;
    out_gate[(size_t)(b0 + b) * 3 + 0] = e0 * inv;
    out_gate[(size_t)(b0 + b) * 3 + 1] = e1 * inv;
    out_gate[(size_t)(b0 + b) * 3 + 2] = e2 * inv;
  }
  __syncthreads();

  // ---------------- Step 5: gated fusion (16B-chunked) -----------------------
  for (int i = tid; i < KBM * 64; i += NTHREADS) {   // 16 rows x 64 chunks of 8
    int b = i >> 6, c = (i & 63) * 8;
    float g0 = sGate[b * 3], g1 = sGate[b * 3 + 1], g2 = sGate[b * 3 + 2];
    const bf16t* ab = sAttO + (b * 3) * 520 + c;
    C8 a0 = lds_c8(ab);
    C8 a1 = lds_c8(ab + 520);
    C8 a2 = lds_c8(ab + 1040);
    C8 o;
    #pragma unroll
    for (int e = 0; e < 8; ++e)
      o.h[e] = (bf16t)(g0 * (float)a0.h[e] + g1 * (float)a1.h[e] +
                       g2 * (float)a2.h[e]);
    *(u32x4*)(sFused + b * 520 + c) = o.u;
  }
  __syncthreads();

  // ---------------- Step 6: classifier ---------------------------------------
  {
    int n0 = wave * 32;
    v8f acc[2] = {vz, vz};
    gemm_rowblock<2>([&](int k0) { return sFused + k0; }, 520,
                     Wc1, 512, n0, 512, acc);
    #pragma unroll
    for (int jj = 0; jj < 2; ++jj) {
      int nc = n0 + jj * 16 + cn;
      float bias = gloadf(bc1 + nc);
      #pragma unroll
      for (int r = 0; r < 8; ++r)
        sTmp[(cm + r) * 516 + nc] = acc[jj][r] + bias;
    }
  }
  __syncthreads();
  {
    // LayerNorm(256) + ReLU, one wave per two rows
    for (int rr = 0; rr < 2; ++rr) {
      int row = wave * 2 + rr;
      float v[8], s1 = 0.f, s2 = 0.f;
      #pragma unroll
      for (int i = 0; i < 8; ++i) {
        float t = sTmp[row * 516 + lane + 32 * i];
        v[i] = t; s1 += t; s2 += t * t;
      }
      #pragma unroll
      for (int m = 16; m >= 1; m >>= 1) {
        s1 += __shfl_xor(s1, m, 32);
        s2 += __shfl_xor(s2, m, 32);
      }
      float mean = s1 * (1.f / 256.f);
      float rstd = rsqrtf(s2 * (1.f / 256.f) - mean * mean + 1e-5f);
      bf16t* dst = sClsH + row * 264;
      #pragma unroll
      for (int i = 0; i < 8; ++i) {
        int e = lane + 32 * i;
        float y = (v[i] - mean) * rstd * gloadf(cg + e) + gloadf(cbeta + e);
        dst[e] = (bf16t)fmaxf(y, 0.f);
      }
    }
  }
  __syncthreads();
  if (tid < KBM * 6) {
    int b = tid / 6, c = tid - 6 * (tid / 6);
    float acc = gloadf(bc2 + c);
    const bf16t* hb = sClsH + b * 264;
    const bf16t* w  = Wc2 + c * 256;
    #pragma unroll 8
    for (int cc = 0; cc < 32; ++cc) {
      C8 hc = lds_c8(hb + cc * 8);
      C8 wc = glb_c8(w + cc * 8);
      #pragma unroll
      for (int e = 0; e < 8; ++e) acc += (float)hc.h[e] * (float)wc.h[e];
    }
    out_logits[(size_t)(b0 + b) * 6 + c] = acc;
  }
}

extern "C" void kernel_launch(void* const* d_in, const int* in_sizes, int n_in,
                              void* d_out, int out_size, void* d_ws, size_t ws_size,
                              hipStream_t stream) {
  (void)in_sizes; (void)n_in; (void)out_size; (void)ws_size;
  bf16t* ws = (bf16t*)d_ws;
  size_t off = 0;
  auto cv = [&](int idx, size_t n) -> bf16t* {
    bf16t* dst = ws + off;
    int n4 = (int)(n / 4);                       // all sizes are multiples of 4
    int blocks = (n4 + NTHREADS - 1) / NTHREADS;
    cvt_bf16<<<blocks, NTHREADS, 0, stream>>>((const float*)d_in[idx], dst, n4);
    off += n;
    return dst;
  };
  bf16t* Wr  = cv(3,  (size_t)512 * 768);
  bf16t* Wp  = cv(7,  (size_t)512 * 640);
  bf16t* Wf  = cv(11, (size_t)512 * 1024);
  bf16t* Win = cv(15, (size_t)1536 * 512);
  bf16t* Wou = cv(17, (size_t)512 * 512);
  bf16t* Wg1 = cv(19, (size_t)512 * 1536);
  bf16t* Wg2 = cv(21, (size_t)3 * 512);
  bf16t* Wc1 = cv(23, (size_t)256 * 512);
  bf16t* Wc2 = cv(27, (size_t)6 * 256);

  float* out_logits = (float*)d_out;
  float* out_gate   = out_logits + (size_t)16384 * 6;

  dim3 grid(16384 / KBM), block(NTHREADS);
  size_t shbytes = 289984;
  fusion_fwd<<<grid, block, shbytes, stream>>>(
      (const float*)d_in[0], (const float*)d_in[1], (const float*)d_in[2],
      Wr,  (const float*)d_in[4],  (const float*)d_in[5],  (const float*)d_in[6],
      Wp,  (const float*)d_in[8],  (const float*)d_in[9],  (const float*)d_in[10],
      Wf,  (const float*)d_in[12], (const float*)d_in[13], (const float*)d_in[14],
      Win, (const float*)d_in[16],
      Wou, (const float*)d_in[18],
      Wg1, (const float*)d_in[20],
      Wg2, (const float*)d_in[22],
      Wc1, (const float*)d_in[24], (const float*)d_in[25], (const float*)d_in[26],
      Wc2, (const float*)d_in[28],
      out_logits, out_gate);
}
